// Cfc_84722524880936
// MI455X (gfx1250) — matrix operand order repaired and checked by reading, not compiled, not run on hardware
//
#include <hip/hip_runtime.h>
#include <cmath>

// Problem dims (fixed by reference)
#define BDIM   64
#define TDIM   1024
#define DDIM   256
#define HDIM   512
#define BUDIM  512
#define K1     (DDIM + HDIM)   // 768

typedef __bf16 bf16;
typedef __attribute__((ext_vector_type(16))) __bf16 v16bf;
typedef __attribute__((ext_vector_type(8)))  __bf16 v8bf;
typedef __attribute__((ext_vector_type(8)))  float  v8f;

// ---------------------------------------------------------------------------
// Fast transcendentals: use gfx1250 hardware TRANS32 ops instead of IEEE
// divide expansions / libm calls. TANH is a native TRANS op on CDNA5.
// ---------------------------------------------------------------------------
__device__ __forceinline__ float fast_rcp(float x) {
    return __builtin_amdgcn_rcpf(x);              // v_rcp_f32
}
__device__ __forceinline__ float fast_sigmoid(float x) {
    return fast_rcp(1.0f + __expf(-x));           // v_exp_f32 + v_rcp_f32
}
__device__ __forceinline__ float fast_tanh(float x) {
#if __has_builtin(__builtin_amdgcn_tanhf)
    return __builtin_amdgcn_tanhf(x);             // v_tanh_f32 (gfx1250)
#else
    return 1.0f - 2.0f * fast_rcp(1.0f + __expf(2.0f * x));
#endif
}

// ---------------------------------------------------------------------------
// Prep kernel: transpose f32 weight (K x N, row major) into bf16 (N x K).
// Makes every WMMA B-fragment a contiguous 32-byte per-lane load.
// ---------------------------------------------------------------------------
__global__ void transpose_to_bf16(const float* __restrict__ src,
                                  bf16* __restrict__ dst, int K, int N) {
    int idx = blockIdx.x * blockDim.x + threadIdx.x;
    if (idx >= K * N) return;
    int n = idx / K;
    int k = idx - n * K;
    dst[idx] = (bf16)src[(size_t)k * N + n];
}

// ---------------------------------------------------------------------------
// Fragment loaders (wave32 WMMA bf16; contiguous K order chosen here, same for A and B)
// A 16x32: lane(l<16)=row l, K in {0..15}; lane(l+16)=row l, K in {16..31}
// B 32x16 (from N-major weight): lane holds column n=lane%16, K = (lane/16)*16 + j
// ---------------------------------------------------------------------------
__device__ __forceinline__ v16bf load_a_frag(const bf16* rowbase, int k0, int half) {
    v8bf lo = *(const v8bf*)(rowbase + k0 + half * 16);
    v8bf hi = *(const v8bf*)(rowbase + k0 + half * 16 + 8);
    v16bf a;
#pragma unroll
    for (int j = 0; j < 8; ++j) { a[j] = lo[j]; a[j + 8] = hi[j]; }
    return a;
}

#define WMMA_BF16(A, B, C) \
    __builtin_amdgcn_wmma_f32_16x16x32_bf16(false, (A), false, (B), (short)0, (C), false, false)

// ---------------------------------------------------------------------------
// Persistent CfC scan kernel.
// grid = B/16 workgroups (one 16-row batch tile each), block = 1024 (32 waves).
// Wave w owns output columns [16w, 16w+16) of every GEMM for all T steps.
// s-state lives in a v8f register fragment for the whole sequence.
// ---------------------------------------------------------------------------
__global__ __launch_bounds__(1024) void cfc_scan_kernel(
    const float* __restrict__ x,   const float* __restrict__ ts,
    const float* __restrict__ h0,  const float* __restrict__ s0,
    const float* __restrict__ bbb, const float* __restrict__ bff1,
    const float* __restrict__ bff2,const float* __restrict__ bta,
    const float* __restrict__ btb,
    const bf16* __restrict__ WbbT,  const bf16* __restrict__ Wff1T,
    const bf16* __restrict__ Wff2T, const bf16* __restrict__ WtaT,
    const bf16* __restrict__ WtbT,
    float* __restrict__ out)
{
    __shared__ bf16  zt[16][K1];      // z = [x_t | h]  (bf16 GEMM operand)
    __shared__ bf16  bbs[16][BUDIM];  // silu backbone activations
    __shared__ float wts_s[16];       // per-row time gate

    const int tid  = threadIdx.x;
    const int wave = tid >> 5;        // 0..31
    const int lane = tid & 31;
    const int half = lane >> 4;       // 0/1: K-half for A/B, M-half for C/D
    const int nl   = lane & 15;       // A row within tile / N column within tile
    const int col  = wave * 16 + nl;  // global output column (0..511)
    const int bbase = blockIdx.x * 16;

    // Loop-invariant per-lane biases
    const float bbbv = bbb[col];
    const float b1v  = bff1[col];
    const float b2v  = bff2[col];
    const float btav = bta[col];
    const float btbv = btb[col];

    // ---- init: h0 -> zt[:,256:768] (bf16), s0 -> register fragment ----
#pragma unroll
    for (int it = 0; it < 2; ++it) {
        int e  = tid + it * 1024;             // 16 rows * 128 float4 groups
        int m  = e >> 7;
        int c4 = (e & 127) * 4;
        const float4 v = *(const float4*)(h0 + (size_t)(bbase + m) * HDIM + c4);
        bf16* p = &zt[m][DDIM + c4];
        p[0] = (bf16)v.x; p[1] = (bf16)v.y; p[2] = (bf16)v.z; p[3] = (bf16)v.w;
    }
    v8f s_frag;
#pragma unroll
    for (int r = 0; r < 8; ++r)
        s_frag[r] = s0[(size_t)(bbase + r + half * 8) * HDIM + col];

    for (int t = 0; t < TDIM; ++t) {
        // ---- stage x_t into zt[:,0:256] (one float4 per thread) ----
        {
            int m  = tid >> 6;                // 16 rows * 64 float4 groups = 1024
            int c4 = (tid & 63) * 4;
            const float* xp = x + ((size_t)(bbase + m) * TDIM + t) * DDIM + c4;
            const float4 v = *(const float4*)xp;
            bf16* p = &zt[m][c4];
            p[0] = (bf16)v.x; p[1] = (bf16)v.y; p[2] = (bf16)v.z; p[3] = (bf16)v.w;
            if (t + 1 < TDIM)                 // warm L2/L0 for next step's x
                __builtin_prefetch(xp + DDIM, 0, 3);
        }
        if (tid < 16) {
            float tn = 1.0f / ts[(size_t)(bbase + tid) * TDIM + t];
            wts_s[tid] = (tn > 0.0f) ? __expf(tn * (1.0f - 2.0f * __logf(tn))) : 0.0f;
        }
        __syncthreads();

        // ---- GEMM 1: bb = silu(z @ Wbb + bbb), K = 768 ----
        v8f acc = {};
#pragma unroll 4
        for (int k0 = 0; k0 < K1; k0 += 32) {
            v16bf a = load_a_frag(&zt[nl][0], k0, half);
            v16bf b = *(const v16bf*)(WbbT + (size_t)col * K1 + k0 + half * 16);
            acc = WMMA_BF16(a, b, acc);
        }
#pragma unroll
        for (int r = 0; r < 8; ++r) {
            float v  = acc[r] + bbbv;
            float sv = v * fast_sigmoid(v);              // silu via v_exp + v_rcp
            bbs[r + half * 8][col] = (bf16)sv;
        }
        __syncthreads();

        // ---- GEMMs 2..5 fused: share each A-fragment across 4 WMMAs, K = 512 ----
        v8f a1 = {}, a2 = {}, aa = {}, ab = {};
#pragma unroll 2
        for (int k0 = 0; k0 < BUDIM; k0 += 32) {
            v16bf a  = load_a_frag(&bbs[nl][0], k0, half);
            const size_t wo = (size_t)col * BUDIM + k0 + half * 16;
            v16bf b1 = *(const v16bf*)(Wff1T + wo);
            v16bf b2 = *(const v16bf*)(Wff2T + wo);
            v16bf b3 = *(const v16bf*)(WtaT  + wo);
            v16bf b4 = *(const v16bf*)(WtbT  + wo);
            a1 = WMMA_BF16(a, b1, a1);
            a2 = WMMA_BF16(a, b2, a2);
            aa = WMMA_BF16(a, b3, aa);
            ab = WMMA_BF16(a, b4, ab);
        }

        // ---- gating epilogue: update s (registers), h -> out + zt for next z ----
#pragma unroll
        for (int r = 0; r < 8; ++r) {
            int   m    = r + half * 8;
            float f1   = fast_tanh(a1[r] + b1v);         // v_tanh_f32
            float f2   = fast_tanh(a2[r] + b2v);
            float tsum = (aa[r] + btav) * wts_s[m] + (ab[r] + btbv);
            float s    = s_frag[r] + tsum;
            s_frag[r]  = s;
            float ti   = fast_sigmoid(s);
            float h    = f1 * (1.0f - ti) + ti * f2;
            out[((size_t)(bbase + m) * TDIM + t) * HDIM + col] = h;
            zt[m][DDIM + col] = (bf16)h;                 // recurrent feed
        }
        __syncthreads();
    }
}

// ---------------------------------------------------------------------------
// Host launcher
// ---------------------------------------------------------------------------
extern "C" void kernel_launch(void* const* d_in, const int* in_sizes, int n_in,
                              void* d_out, int out_size, void* d_ws, size_t ws_size,
                              hipStream_t stream) {
    (void)in_sizes; (void)n_in; (void)out_size; (void)ws_size;
    const float* x    = (const float*)d_in[0];
    const float* ts   = (const float*)d_in[1];
    const float* h0   = (const float*)d_in[2];
    const float* s0   = (const float*)d_in[3];
    const float* Wbb  = (const float*)d_in[4];
    const float* bbb  = (const float*)d_in[5];
    const float* Wff1 = (const float*)d_in[6];
    const float* bff1 = (const float*)d_in[7];
    const float* Wff2 = (const float*)d_in[8];
    const float* bff2 = (const float*)d_in[9];
    const float* Wta  = (const float*)d_in[10];
    const float* bta  = (const float*)d_in[11];
    const float* Wtb  = (const float*)d_in[12];
    const float* btb  = (const float*)d_in[13];
    float* out = (float*)d_out;

    // Workspace: transposed bf16 weights, N-major so B-fragments are contiguous.
    bf16* WbbT  = (bf16*)d_ws;
    bf16* Wff1T = WbbT  + (size_t)BUDIM * K1;
    bf16* Wff2T = Wff1T + (size_t)HDIM * BUDIM;
    bf16* WtaT  = Wff2T + (size_t)HDIM * BUDIM;
    bf16* WtbT  = WtaT  + (size_t)HDIM * BUDIM;

    const int TB = 256;
    transpose_to_bf16<<<(K1 * BUDIM + TB - 1) / TB, TB, 0, stream>>>(Wbb,  WbbT,  K1,    BUDIM);
    transpose_to_bf16<<<(BUDIM * HDIM + TB - 1) / TB, TB, 0, stream>>>(Wff1, Wff1T, BUDIM, HDIM);
    transpose_to_bf16<<<(BUDIM * HDIM + TB - 1) / TB, TB, 0, stream>>>(Wff2, Wff2T, BUDIM, HDIM);
    transpose_to_bf16<<<(BUDIM * HDIM + TB - 1) / TB, TB, 0, stream>>>(Wta,  WtaT,  BUDIM, HDIM);
    transpose_to_bf16<<<(BUDIM * HDIM + TB - 1) / TB, TB, 0, stream>>>(Wtb,  WtbT,  BUDIM, HDIM);

    // One persistent workgroup per 16-row batch tile; 32 waves each.
    cfc_scan_kernel<<<BDIM / 16, 1024, 0, stream>>>(
        x, ts, h0, s0, bbb, bff1, bff2, bta, btb,
        WbbT, Wff1T, Wff2T, WtaT, WtbT, out);
}
